// GRUDecoder_13082470383986
// MI455X (gfx1250) — matrix operand order repaired and checked by reading, not compiled, not run on hardware
//
#include <hip/hip_runtime.h>
#include <hip/hip_bf16.h>
#include <math.h>

// ---------------------------------------------------------------------------
// Fused GRU decoder for MI455X (gfx1250, wave32, WMMA).
//   grid  = B/16 workgroups, each owns 16 batch rows for all 512 steps
//   block = 128 threads = 4 waves; wave w owns gh n-tiles [3w..3w+2] (N=192)
//           and head n-tiles [5w..5w+4] (N padded 274->320)
//   W_hh^T B-fragments are register-resident (hot: used every step).
//   Head B-fragments live in LDS, pre-swizzled into WMMA-B lane layout.
//   h lives in LDS: f16 copy (WMMA A operand) + f32 shadow (gating math).
//   Gating: fast TRANS-unit sigmoid/tanh; b_hh r/z folded into gi at init,
//   b_hh n-component register-hoisted -> no loop-invariant LDS reads.
//   Head stores use 32-bit byte offsets from the uniform out base (whole
//   output < 2^31 bytes) -> saddr-form global stores, less address state.
//   Heads are fused per step -> hs (0.5 GB round trip) never touches HBM.
// ---------------------------------------------------------------------------

typedef __attribute__((ext_vector_type(16))) _Float16 v16h;
typedef __attribute__((ext_vector_type(8)))  _Float16 v8h;
typedef __attribute__((ext_vector_type(8)))  float    v8f;

#define HDIM 64
#define LDIM 128
#define GIS  200   // gi LDS row stride (floats)
#define GHS  200   // gh LDS row stride (floats)
#define H16S 72    // f16 h LDS row stride (halves) -> conflict-skewed, 16B aligned

__device__ __forceinline__ v8f wmma16(v16h a, v16h b, v8f c) {
  // emits v_wmma_f32_16x16x32_f16
  return __builtin_amdgcn_wmma_f32_16x16x32_f16(false, a, false, b,
                                                (short)0, c, false, false);
}

// fast sigmoid/tanh on the TRANS unit: one v_exp_f32 + one v_rcp_f32 each
__device__ __forceinline__ float fast_sigmoid(float x) {
  return __builtin_amdgcn_rcpf(1.f + __expf(-x));
}
__device__ __forceinline__ float fast_tanh(float x) {
  return 2.f * __builtin_amdgcn_rcpf(1.f + __expf(-2.f * x)) - 1.f;
}

// A fragment 16x32 (this code's contiguous k order, same as B): lanes 0-15 hold row M=lane,
//   halves 0..15 -> K = 32*kt + 16*hi + i     (hi = lane>=16)
__device__ __forceinline__ v16h load_a_frag(const _Float16* h16, int kt,
                                            int l, int hi) {
  const _Float16* p = h16 + l * H16S + kt * 32 + hi * 16;
  v8h lo = *(const v8h*)p;          // 16B ds_load
  v8h hh = *(const v8h*)(p + 8);    // 16B ds_load
  v16h a;
#pragma unroll
  for (int i = 0; i < 8; ++i) { a[i] = lo[i]; a[8 + i] = hh[i]; }
  return a;
}

__global__ void __launch_bounds__(128)
gru_dec_kernel(const float* __restrict__ latent,
               const float* __restrict__ W_lh,   const float* __restrict__ b_lh,
               const float* __restrict__ W_ih,   const float* __restrict__ b_ih,
               const float* __restrict__ W_hh,   const float* __restrict__ b_hh,
               const float* __restrict__ W_out,  const float* __restrict__ b_out,
               const float* __restrict__ W_sport,const float* __restrict__ b_sport,
               const float* __restrict__ W_dport,const float* __restrict__ b_dport,
               const float* __restrict__ W_proto,const float* __restrict__ b_proto,
               const int* __restrict__ seq_len,
               float* __restrict__ out, int Btot)
{
  __shared__ float gil[16 * GIS];                 // gi (+b_hh r/z) (16 x 192, padded)
  __shared__ float ghl[16 * GHS];                 // gh (16 x 192, padded); reused as latent stage
  __shared__ float hf32[16 * HDIM];               // f32 shadow of h
  __shared__ __align__(16) _Float16 h16s[16 * H16S]; // f16 h (WMMA A source)
  // Head weight fragments, pre-swizzled into WMMA-B lane layout:
  // [wave][tile j][kt][lane][16 halves] -> per-lane contiguous 32B
  __shared__ __align__(16) _Float16 hfrag[4][5][2][32][16];   // 40 KB

  const int t    = threadIdx.x;
  const int w    = t >> 5;        // wave id 0..3
  const int lane = t & 31;
  const int l    = lane & 15;
  const int hi   = lane >> 4;
  const int rowBase = blockIdx.x * 16;
  const int S = seq_len[0];
  const int gm  = t >> 3;
  const int gc0 = (t & 7) * 8;

  // element offsets of the four heads inside the flat out buffer
  const size_t e_recon = 0;
  const size_t e_sport = e_recon + (size_t)Btot * S * 10;
  const size_t e_dport = e_sport + (size_t)Btot * S * 128;
  const size_t e_proto = e_dport + (size_t)Btot * S * 128;

  // ---------------- init: stage latent (overlaid in ghl) ----------
  float* latl = ghl;
  for (int i = t; i < 16 * LDIM; i += 128)
    latl[i] = latent[(size_t)rowBase * LDIM + i];
  __syncthreads();

  // h0 = tanh(latent @ W_lh^T + b_lh): 8 outputs/thread
  {
#pragma unroll
    for (int i = 0; i < 8; ++i) {
      int c = gc0 + i;
      float acc = b_lh[c];
      for (int k = 0; k < LDIM; ++k)
        acc += latl[gm * LDIM + k] * W_lh[c * LDIM + k];
      float h0 = fast_tanh(acc);
      hf32[gm * HDIM + c] = h0;
      h16s[gm * H16S + c] = (_Float16)h0;
    }
  }
  __syncthreads();

  // gi = h0 @ W_ih^T + b_ih (+ b_hh for the additive r/z gates)
  for (int o = t * 24; o < t * 24 + 24; ++o) {
    int m = o / 192, c = o % 192;
    float acc = b_ih[c] + (c < 128 ? b_hh[c] : 0.f);
    for (int k = 0; k < HDIM; ++k)
      acc += hf32[m * HDIM + k] * W_ih[c * HDIM + k];
    gil[m * GIS + c] = acc;
  }

  // b_hh n-component (multiplied by r in the PyTorch form) -> registers
  float bhn[8];
#pragma unroll
  for (int i = 0; i < 8; ++i) bhn[i] = b_hh[128 + gc0 + i];

  // ---- register-resident B fragments for W_hh^T (contiguous k order, matching A:
  //      lane = N, half h -> K = 32*kt + 16*hi + h), gathered once via L2 ----
  v16h bhhf[3][2];
#pragma unroll
  for (int j = 0; j < 3; ++j) {
    int n = (3 * w + j) * 16 + l;
#pragma unroll
    for (int kt = 0; kt < 2; ++kt) {
      int kb = kt * 32 + hi * 16;
      v16h b;
#pragma unroll
      for (int h = 0; h < 16; ++h) b[h] = (_Float16)W_hh[n * HDIM + kb + h];
      bhhf[j][kt] = b;
    }
  }

  // ---- head weight fragments: gather once, park in LDS in B layout ----
  float hbreg[5];
  unsigned boff[5], rstr[5], binc[5];
  bool hval[5];
#pragma unroll
  for (int j = 0; j < 5; ++j) {
    int c = (5 * w + j) * 16 + l;
#pragma unroll
    for (int kt = 0; kt < 2; ++kt) {
      int kb = kt * 32 + hi * 16;
      v16h b;
#pragma unroll
      for (int h = 0; h < 16; ++h) {
        int k = kb + h;
        float v = 0.f;
        if (c < 10)       v = W_out [c * HDIM + k];
        else if (c < 138) v = W_sport[(c - 10) * HDIM + k];
        else if (c < 266) v = W_dport[(c - 138) * HDIM + k];
        else if (c < 274) v = W_proto[(c - 266) * HDIM + k];
        b[h] = (_Float16)v;
      }
      *(v16h*)&hfrag[w][j][kt][lane][0] = b;   // 32B per-lane contiguous
    }
    size_t ebase; int cc, dm;
    if (c < 10)       { ebase = e_recon; cc = c;        dm = 10;  hval[j] = true;  hbreg[j] = b_out[c]; }
    else if (c < 138) { ebase = e_sport; cc = c - 10;   dm = 128; hval[j] = true;  hbreg[j] = b_sport[c - 10]; }
    else if (c < 266) { ebase = e_dport; cc = c - 138;  dm = 128; hval[j] = true;  hbreg[j] = b_dport[c - 138]; }
    else if (c < 274) { ebase = e_proto; cc = c - 266;  dm = 8;   hval[j] = true;  hbreg[j] = b_proto[c - 266]; }
    else              { ebase = 0;       cc = 0;        dm = 0;   hval[j] = false; hbreg[j] = 0.f; }
    // whole output < 2^31 bytes -> 32-bit byte offsets from uniform base
    size_t e0 = ebase + ((size_t)(rowBase + 8 * hi) * S) * dm + cc;
    boff[j] = (unsigned)(e0 * 4u);
    rstr[j] = (unsigned)((size_t)S * dm * 4u);   // row stride (bytes)
    binc[j] = (unsigned)(dm * 4u);               // per-step advance (bytes)
  }
  __syncthreads();

  v16h a0 = load_a_frag(h16s, 0, l, hi);
  v16h a1 = load_a_frag(h16s, 1, l, hi);

  // ----------------------------- time loop --------------------------------
  for (int s = 0; s < S; ++s) {
    // gh = h @ W_hh^T for my 3 n-tiles (6 WMMAs) -> LDS
#pragma unroll
    for (int j = 0; j < 3; ++j) {
      v8f acc = {0.f, 0.f, 0.f, 0.f, 0.f, 0.f, 0.f, 0.f};
      acc = wmma16(a0, bhhf[j][0], acc);
      acc = wmma16(a1, bhhf[j][1], acc);
      int nb = (3 * w + j) * 16 + l;
#pragma unroll
      for (int r = 0; r < 8; ++r) ghl[(r + 8 * hi) * GHS + nb] = acc[r];
    }
    __syncthreads();

    // gating (PyTorch form: n = tanh(gi_n + r*(gh_n + b_hh_n)))
#pragma unroll
    for (int i = 0; i < 8; ++i) {
      int c = gc0 + i;
      float ghr = ghl[gm * GHS + c];
      float ghz = ghl[gm * GHS + 64 + c];
      float ghn = ghl[gm * GHS + 128 + c] + bhn[i];
      float r = fast_sigmoid(gil[gm * GIS + c]      + ghr);
      float z = fast_sigmoid(gil[gm * GIS + 64 + c] + ghz);
      float n = fast_tanh(gil[gm * GIS + 128 + c] + r * ghn);
      float hp = hf32[gm * HDIM + c];
      float hn = (1.f - z) * n + z * hp;
      hf32[gm * HDIM + c] = hn;
      h16s[gm * H16S + c] = (_Float16)hn;
    }
    __syncthreads();

    // reload A = h_new; shared between heads (this step) and gh (next step)
    a0 = load_a_frag(h16s, 0, l, hi);
    a1 = load_a_frag(h16s, 1, l, hi);

    // heads on h_new (10 WMMAs), B frags streamed from LDS, bias-seeded,
    // results written straight to HBM (coalesced saddr-form dword stores)
#pragma unroll
    for (int j = 0; j < 5; ++j) {
      v16h b0 = *(const v16h*)&hfrag[w][j][0][lane][0];
      v16h b1 = *(const v16h*)&hfrag[w][j][1][lane][0];
      v8f acc;
#pragma unroll
      for (int r = 0; r < 8; ++r) acc[r] = hbreg[j];
      acc = wmma16(a0, b0, acc);
      acc = wmma16(a1, b1, acc);
      if (hval[j]) {
#pragma unroll
        for (int r = 0; r < 8; ++r)
          *(float*)((char*)out + (boff[j] + (unsigned)r * rstr[j])) = acc[r];
      }
      boff[j] += binc[j];
    }
  }
}

extern "C" void kernel_launch(void* const* d_in, const int* in_sizes, int n_in,
                              void* d_out, int out_size, void* d_ws, size_t ws_size,
                              hipStream_t stream) {
  const float* latent  = (const float*)d_in[0];
  const float* W_lh    = (const float*)d_in[1];
  const float* b_lh    = (const float*)d_in[2];
  const float* W_ih    = (const float*)d_in[3];
  const float* b_ih    = (const float*)d_in[4];
  const float* W_hh    = (const float*)d_in[5];
  const float* b_hh    = (const float*)d_in[6];
  const float* W_out   = (const float*)d_in[7];
  const float* b_out   = (const float*)d_in[8];
  const float* W_sport = (const float*)d_in[9];
  const float* b_sport = (const float*)d_in[10];
  const float* W_dport = (const float*)d_in[11];
  const float* b_dport = (const float*)d_in[12];
  const float* W_proto = (const float*)d_in[13];
  const float* b_proto = (const float*)d_in[14];
  const int*   seq     = (const int*)d_in[15];

  int Btot = in_sizes[0] / 128;     // latent is (B, 128)
  dim3 grid(Btot / 16), block(128);
  hipLaunchKernelGGL(gru_dec_kernel, grid, block, 0, stream,
                     latent, W_lh, b_lh, W_ih, b_ih, W_hh, b_hh,
                     W_out, b_out, W_sport, b_sport, W_dport, b_dport,
                     W_proto, b_proto, seq, (float*)d_out, Btot);
}
